// LSTM_Model_44641890074838
// MI455X (gfx1250) — compile-verified, matrix operand order checked by reading, not run on hardware
//
#include <hip/hip_runtime.h>

typedef __bf16 bhalf;
typedef __attribute__((ext_vector_type(16))) __bf16 v16bf;
typedef __attribute__((ext_vector_type(8)))  float v8f;
typedef __attribute__((ext_vector_type(4)))  unsigned int v4u;

#define BB 32
#define TT 512
#define HH 128
#define G4 512   // 4*H

#define LOG2E 1.4426950408889634f

// ---- fast activations on CDNA5 hardware transcendentals (no branchy libm) ----
__device__ __forceinline__ float fast_exp(float x) {
  return __builtin_amdgcn_exp2f(x * LOG2E);           // v_exp_f32
}
__device__ __forceinline__ float fast_tanh(float x) {
#if __has_builtin(__builtin_amdgcn_tanhf)
  return __builtin_amdgcn_tanhf(x);                   // v_tanh_f32
#elif __has_builtin(__builtin_amdgcn_tanh_f32)
  return __builtin_amdgcn_tanh_f32(x);
#else
  float e = __builtin_amdgcn_exp2f(2.0f * LOG2E * x);
  return 1.0f - 2.0f * __builtin_amdgcn_rcpf(e + 1.0f);
#endif
}
__device__ __forceinline__ float fast_sigmoid(float x) {
  // exact identity: sigmoid(x) = 0.5*tanh(0.5x)+0.5 -> v_mul + v_tanh + v_fma
  return __builtin_fmaf(0.5f, fast_tanh(0.5f * x), 0.5f);
}

// Load a 16x32 bf16 WMMA fragment from a row-major [rows][ld] bf16 buffer.
// Per CDNA5 ISA 16-bit A/B striping: lane l -> row (l&15), K-base (l>=16 ? 8:0),
// vector elements 0..7 = K[kb..kb+7], elements 8..15 = K[kb+16..kb+23].
// Works for A ([M][K]) and for B when B^T is stored row-major as [N][K].
__device__ __forceinline__ v16bf load_frag(const bhalf* __restrict__ base, int row0, int ld, int k0) {
  const int lane = threadIdx.x & 31;
  const int r  = row0 + (lane & 15);
  const int kb = k0 + ((lane >> 4) << 3);
  const bhalf* p = base + (size_t)r * ld + kb;
  union { v4u u[2]; v16bf v; } t;
  t.u[0] = *(const v4u*)(p);        // 8 contiguous bf16
  t.u[1] = *(const v4u*)(p + 16);   // 8 contiguous bf16, K+16
  return t.v;
}

__device__ __forceinline__ v8f wmma_bf16(v16bf a, v16bf b, v8f c) {
  // (neg_a, A, neg_b, B, c_mod, C, reuse_a, reuse_b)
  return __builtin_amdgcn_wmma_f32_16x16x32_bf16(false, a, false, b, (short)0, c, false, false);
}

// ---------------------------------------------------------------- prep kernels
__global__ void zero_f32(float* __restrict__ p, size_t n) {
  for (size_t i = blockIdx.x * (size_t)blockDim.x + threadIdx.x; i < n;
       i += (size_t)gridDim.x * blockDim.x) p[i] = 0.0f;
}

__global__ void cvt_bf16(const float* __restrict__ src, bhalf* __restrict__ dst, size_t n) {
  for (size_t i = blockIdx.x * (size_t)blockDim.x + threadIdx.x; i < n;
       i += (size_t)gridDim.x * blockDim.x) dst[i] = (bhalf)src[i];
}

// dst[m][n][k] = src[m][k][n]  (build B^T row-major [N][K] in bf16)
__global__ void transpose_bf16(const float* __restrict__ src, bhalf* __restrict__ dst,
                               int rowsK, int colsN, int nmat) {
  size_t total = (size_t)nmat * rowsK * colsN;
  for (size_t i = blockIdx.x * (size_t)blockDim.x + threadIdx.x; i < total;
       i += (size_t)gridDim.x * blockDim.x) {
    size_t m   = i / ((size_t)rowsK * colsN);
    size_t rem = i - m * (size_t)rowsK * colsN;
    size_t n   = rem / rowsK;
    size_t k   = rem - n * (size_t)rowsK;
    dst[i] = (bhalf)src[(m * rowsK + k) * colsN + n];
  }
}

// ---------------------------------------------------------------- stage 1
// One persistent workgroup (8 wave32) per input stream; 512-step LSTM scan.
// Per step: gates[32,512] = x*Wih^T + bias + h@Whh^T via 32 WMMA per wave.
__global__ __launch_bounds__(256) void stage1_lstm(
    const float* __restrict__ Y, const float* __restrict__ Xp,
    const float* __restrict__ Xn, const float* __restrict__ Xi,
    const float* __restrict__ Wih, const float* __restrict__ bih,
    const float* __restrict__ bhh, const bhalf* __restrict__ WhhT,
    float* __restrict__ result) {
  __shared__ alignas(16) bhalf sh[BB * HH];  // h state, bf16 (WMMA A)
  __shared__ float sc[BB * HH];              // c state
  __shared__ float sg[BB * G4];              // activated gates
  __shared__ float sx[BB];
  __shared__ float sWih[G4];
  __shared__ float sbias[G4];

  const int tid  = threadIdx.x;
  const int wave = tid >> 5;
  const int lane = tid & 31;

  const int s = blockIdx.x;               // stream id 0..33
  const float* xbase; size_t bstride; int colOff; float scale; bool useAtomic;
  if (s == 0)      { xbase = Y;                          bstride = TT;    colOff = 0;      scale = 1.0f;       useAtomic = false; }
  else if (s == 1) { xbase = Xi;                         bstride = TT;    colOff = 3 * HH; scale = 1.0f;       useAtomic = false; }
  else if (s < 18) { xbase = Xp + (size_t)(s - 2)  * TT; bstride = 16*TT; colOff = HH;     scale = 1.0f/16.0f; useAtomic = true; }
  else             { xbase = Xn + (size_t)(s - 18) * TT; bstride = 16*TT; colOff = 2 * HH; scale = 1.0f/16.0f; useAtomic = true; }

  for (int i = tid; i < G4; i += 256) { sWih[i] = Wih[i]; sbias[i] = bih[i] + bhh[i]; }
  for (int i = tid; i < BB * HH; i += 256) { sh[i] = (bhalf)0.0f; sc[i] = 0.0f; }
  __syncthreads();

  for (int t = 0; t < TT; ++t) {
    if (tid < BB) sx[tid] = xbase[(size_t)tid * bstride + t];
    __syncthreads();

    for (int mt = 0; mt < 2; ++mt) {
      v16bf afr[4];
      #pragma unroll
      for (int kk = 0; kk < 4; ++kk) afr[kk] = load_frag(sh, mt * 16, HH, kk * 32);
      const int mm0 = mt * 16 + ((lane >> 4) << 3);
      #pragma unroll
      for (int ntl = 0; ntl < 4; ++ntl) {
        const int n0 = wave * 64 + ntl * 16;
        const int n  = n0 + (lane & 15);
        v8f acc;
        #pragma unroll
        for (int r = 0; r < 8; ++r) acc[r] = sx[mm0 + r] * sWih[n] + sbias[n];
        #pragma unroll
        for (int kk = 0; kk < 4; ++kk)
          acc = wmma_bf16(afr[kk], load_frag(WhhT, n0, HH, kk * 32), acc);
        // candidate gate -> tanh; others -> sigmoid.  n0 is wave-uniform, so
        // keep the branch OUTSIDE the element loop (single s_cbranch, no
        // per-element exec-mask toggling).
        if ((n0 >= 2 * HH) && (n0 < 3 * HH)) {
          #pragma unroll
          for (int r = 0; r < 8; ++r) sg[(mm0 + r) * G4 + n] = fast_tanh(acc[r]);
        } else {
          #pragma unroll
          for (int r = 0; r < 8; ++r) sg[(mm0 + r) * G4 + n] = fast_sigmoid(acc[r]);
        }
      }
    }
    __syncthreads();

    for (int i = tid; i < BB * HH; i += 256) {
      const int b = i >> 7, j = i & (HH - 1);
      const float ig = sg[b * G4 + j];
      const float fg = sg[b * G4 + HH + j];
      const float gg = sg[b * G4 + 2 * HH + j];
      const float og = sg[b * G4 + 3 * HH + j];
      const float c  = fg * sc[i] + ig * gg;
      const float h  = og * fast_tanh(c);
      sc[i] = c;
      sh[i] = (bhalf)h;
      float* dst = result + ((size_t)b * TT + t) * G4 + colOff + j;
      if (useAtomic) atomicAdd(dst, h * scale);
      else           *dst = h;
    }
    __syncthreads();
  }
}

// ---------------------------------------------------------------- stage 2
// Single persistent workgroup runs the serial Mi-LSTM scan. 112 WMMA tile-jobs
// per step spread over 8 waves: 64 cand/ig jobs, 32 f/o jobs, 16 c@Wa jobs.
__global__ __launch_bounds__(256) void stage2_milstm(
    const float* __restrict__ result,
    const float* __restrict__ init_c, const float* __restrict__ init_h,
    const bhalf* __restrict__ WcT, const bhalf* __restrict__ UcT,
    const bhalf* __restrict__ WgT, const bhalf* __restrict__ UgT,
    const bhalf* __restrict__ WfT, const bhalf* __restrict__ UfT,
    const bhalf* __restrict__ WoT, const bhalf* __restrict__ UoT,
    const bhalf* __restrict__ WaT,
    const float* __restrict__ bc, const float* __restrict__ bg,
    const float* __restrict__ bfv, const float* __restrict__ bov,
    float* __restrict__ Y2) {
  __shared__ alignas(16) bhalf sxt[BB * G4];    // xt, bf16 (WMMA A)
  __shared__ alignas(16) bhalf sh[BB * HH];     // h, bf16
  __shared__ alignas(16) bhalf scb[BB * HH];    // c, bf16 (for c@Wa)
  __shared__ float sc[BB * HH];                 // c, f32
  __shared__ alignas(16) bhalf sl[4 * BB * HH]; // l = ig*cand
  __shared__ float sfo[2 * BB * HH];            // sigmoid(f), sigmoid(o)
  __shared__ float scw[BB * HH];                // c @ Wa
  __shared__ float se[BB * 4];
  __shared__ float sa[BB * 4];
  __shared__ float sbc[4 * HH], sbg[4 * HH], sbf[HH], sbo[HH];

  const int tid  = threadIdx.x;
  const int wave = tid >> 5;
  const int lane = tid & 31;

  for (int i = tid; i < 4 * HH; i += 256) { sbc[i] = bc[i]; sbg[i] = bg[i]; }
  for (int i = tid; i < HH; i += 256)     { sbf[i] = bfv[i]; sbo[i] = bov[i]; }
  for (int i = tid; i < BB * HH; i += 256) {
    const float ci = init_c[i];
    sc[i] = ci; scb[i] = (bhalf)ci; sh[i] = (bhalf)init_h[i];
  }
  __syncthreads();

  for (int t = 0; t < TT; ++t) {
    // stage xt into LDS as bf16
    for (int i = tid; i < BB * G4; i += 256) {
      const int b = i >> 9, col = i & (G4 - 1);
      sxt[i] = (bhalf)result[((size_t)b * TT + t) * G4 + col];
    }
    if (t + 1 < TT)
      for (int b = tid; b < BB; b += 256)
        __builtin_prefetch(result + ((size_t)b * TT + (t + 1)) * G4, 0, 1);
    __syncthreads();

    for (int job = wave; job < 112; job += 8) {
      if (job < 64) {                       // cand/ig for stream k, tile (mt,nt)
        const int k = job >> 4, mt = (job >> 3) & 1, nt = job & 7;
        const int n0 = nt * 16, m0r = mt * 16;
        const int n = n0 + (lane & 15);
        const int mm0 = m0r + ((lane >> 4) << 3);
        v8f ac, ag;
        #pragma unroll
        for (int r = 0; r < 8; ++r) { ac[r] = sbc[k * HH + n]; ag[r] = sbg[k * HH + n]; }
        const bhalf* WcK = WcT + (size_t)k * HH * HH;
        const bhalf* WgK = WgT + (size_t)k * HH * HH;
        const bhalf* UcK = UcT + (size_t)k * HH * HH;
        const bhalf* UgK = UgT + (size_t)k * HH * HH;
        #pragma unroll
        for (int kk = 0; kk < 4; ++kk) {    // xs_k slice of xt
          v16bf a = load_frag(sxt, m0r, G4, k * HH + kk * 32);
          ac = wmma_bf16(a, load_frag(WcK, n0, HH, kk * 32), ac);
          ag = wmma_bf16(a, load_frag(WgK, n0, HH, kk * 32), ag);
        }
        #pragma unroll
        for (int kk = 0; kk < 4; ++kk) {    // h recurrent term
          v16bf a = load_frag(sh, m0r, HH, kk * 32);
          ac = wmma_bf16(a, load_frag(UcK, n0, HH, kk * 32), ac);
          ag = wmma_bf16(a, load_frag(UgK, n0, HH, kk * 32), ag);
        }
        #pragma unroll
        for (int r = 0; r < 8; ++r)
          sl[(k * BB + mm0 + r) * HH + n] = (bhalf)(fast_sigmoid(ag[r]) * fast_tanh(ac[r]));
      } else if (job < 96) {                // f / o gates, K = 512 + 128
        const int j2 = job - 64;
        const int which = j2 >> 4, mt = (j2 >> 3) & 1, nt = j2 & 7;
        const int n0 = nt * 16, m0r = mt * 16;
        const int n = n0 + (lane & 15);
        const int mm0 = m0r + ((lane >> 4) << 3);
        const bhalf* WT = which ? WoT : WfT;
        const bhalf* UT = which ? UoT : UfT;
        const float* bb = which ? sbo : sbf;
        v8f acc;
        #pragma unroll
        for (int r = 0; r < 8; ++r) acc[r] = bb[n];
        #pragma unroll
        for (int kk = 0; kk < 16; ++kk)
          acc = wmma_bf16(load_frag(sxt, m0r, G4, kk * 32),
                          load_frag(WT, n0, G4, kk * 32), acc);
        #pragma unroll
        for (int kk = 0; kk < 4; ++kk)
          acc = wmma_bf16(load_frag(sh, m0r, HH, kk * 32),
                          load_frag(UT, n0, HH, kk * 32), acc);
        #pragma unroll
        for (int r = 0; r < 8; ++r)
          sfo[which * BB * HH + (mm0 + r) * HH + n] = fast_sigmoid(acc[r]);
      } else {                              // c @ Wa (previous c)
        const int j3 = job - 96;
        const int mt = (j3 >> 3) & 1, nt = j3 & 7;
        const int n0 = nt * 16, m0r = mt * 16;
        const int n = n0 + (lane & 15);
        const int mm0 = m0r + ((lane >> 4) << 3);
        v8f acc;
        #pragma unroll
        for (int r = 0; r < 8; ++r) acc[r] = 0.0f;
        #pragma unroll
        for (int kk = 0; kk < 4; ++kk)
          acc = wmma_bf16(load_frag(scb, m0r, HH, kk * 32),
                          load_frag(WaT, n0, HH, kk * 32), acc);
        #pragma unroll
        for (int r = 0; r < 8; ++r) scw[(mm0 + r) * HH + n] = acc[r];
      }
    }
    __syncthreads();

    // attention logits: e[b,k] = <l[b,k,:], (c@Wa)[b,:]>
    if (tid < BB * 4) {
      const int b = tid >> 2, k = tid & 3;
      float d = 0.0f;
      for (int j = 0; j < HH; ++j)
        d += (float)sl[(k * BB + b) * HH + j] * scw[b * HH + j];
      se[tid] = d;
    }
    __syncthreads();
    if (tid < BB) {
      const float e0 = se[tid*4], e1 = se[tid*4+1], e2 = se[tid*4+2], e3 = se[tid*4+3];
      const float m = fmaxf(fmaxf(e0, e1), fmaxf(e2, e3));
      const float x0 = fast_exp(e0 - m), x1 = fast_exp(e1 - m);
      const float x2 = fast_exp(e2 - m), x3 = fast_exp(e3 - m);
      const float rz = __builtin_amdgcn_rcpf(x0 + x1 + x2 + x3);
      sa[tid*4] = x0*rz; sa[tid*4+1] = x1*rz; sa[tid*4+2] = x2*rz; sa[tid*4+3] = x3*rz;
    }
    __syncthreads();

    // state update
    for (int i = tid; i < BB * HH; i += 256) {
      const int b = i >> 7, j = i & (HH - 1);
      float lmix = 0.0f;
      #pragma unroll
      for (int k = 0; k < 4; ++k)
        lmix += sa[b * 4 + k] * (float)sl[(k * BB + b) * HH + j];
      const float c = sfo[i] * sc[i] + lmix;
      const float h = sfo[BB * HH + i] * fast_tanh(c);
      sc[i] = c; scb[i] = (bhalf)c; sh[i] = (bhalf)h;
      Y2[((size_t)b * TT + t) * HH + j] = h;
    }
    __syncthreads();
  }
}

// ---------------------------------------------------------------- stage 3
__global__ __launch_bounds__(128) void stage3_scores(
    const float* __restrict__ Y2, const float* __restrict__ Watt,
    const float* __restrict__ batt, const float* __restrict__ vatt,
    float* __restrict__ scores) {
  const int t = blockIdx.x, b = blockIdx.y, j = threadIdx.x;
  __shared__ float sy[HH];
  __shared__ float red[HH];
  sy[j] = Y2[((size_t)b * TT + t) * HH + j];
  __syncthreads();
  float acc = batt[j];
  for (int k = 0; k < HH; ++k) acc += sy[k] * Watt[k * HH + j];
  red[j] = fast_tanh(acc) * vatt[j];
  __syncthreads();
  for (int s2 = 64; s2 > 0; s2 >>= 1) {
    if (j < s2) red[j] += red[j + s2];
    __syncthreads();
  }
  if (j == 0) scores[(size_t)b * TT + t] = red[0];
}

__global__ __launch_bounds__(128) void stage3_head(
    const float* __restrict__ scores, const float* __restrict__ Y2,
    const float* __restrict__ W1, const float* __restrict__ b1,
    const float* __restrict__ W2, const float* __restrict__ b2,
    const float* __restrict__ W3, const float* __restrict__ b3,
    float* __restrict__ out) {
  const int b = blockIdx.x, j = threadIdx.x;
  __shared__ float red[HH];
  __shared__ float sY3[HH], sh1[HH], sh2[HH];
  float m = -3.0e38f;
  for (int t2 = j; t2 < TT; t2 += HH) m = fmaxf(m, scores[(size_t)b * TT + t2]);
  red[j] = m; __syncthreads();
  for (int s2 = 64; s2 > 0; s2 >>= 1) { if (j < s2) red[j] = fmaxf(red[j], red[j + s2]); __syncthreads(); }
  m = red[0]; __syncthreads();
  float z = 0.0f;
  for (int t2 = j; t2 < TT; t2 += HH) z += fast_exp(scores[(size_t)b * TT + t2] - m);
  red[j] = z; __syncthreads();
  for (int s2 = 64; s2 > 0; s2 >>= 1) { if (j < s2) red[j] += red[j + s2]; __syncthreads(); }
  z = red[0]; __syncthreads();
  float acc = 0.0f;
  for (int t2 = 0; t2 < TT; ++t2)
    acc += fast_exp(scores[(size_t)b * TT + t2] - m) * Y2[((size_t)b * TT + t2) * HH + j];
  sY3[j] = acc / z;
  __syncthreads();
  float a1 = b1[j];
  for (int k = 0; k < HH; ++k) a1 += sY3[k] * W1[k * HH + j];
  sh1[j] = fmaxf(a1, 0.0f);
  __syncthreads();
  float a2 = b2[j];
  for (int k = 0; k < HH; ++k) a2 += sh1[k] * W2[k * HH + j];
  sh2[j] = fmaxf(a2, 0.0f);
  __syncthreads();
  red[j] = sh2[j] * W3[j];
  __syncthreads();
  for (int s2 = 64; s2 > 0; s2 >>= 1) { if (j < s2) red[j] += red[j + s2]; __syncthreads(); }
  if (j == 0) out[b] = red[0] + b3[0];
}

__global__ void loss_kernel(const float* __restrict__ Target, float* __restrict__ out) {
  __shared__ float sq[BB];
  const int i = threadIdx.x;
  const float d = out[i] - Target[i];
  sq[i] = d * d;
  __syncthreads();
  if (i == 0) {
    float acc = 0.0f;
    for (int k = 0; k < BB; ++k) acc += sq[k];
    out[BB] = acc / (float)BB;
  }
}

// ---------------------------------------------------------------- launcher
extern "C" void kernel_launch(void* const* d_in, const int* in_sizes, int n_in,
                              void* d_out, int out_size, void* d_ws, size_t ws_size,
                              hipStream_t stream) {
  (void)in_sizes; (void)n_in; (void)out_size; (void)ws_size;
  const float* Y      = (const float*)d_in[0];
  const float* Xp     = (const float*)d_in[1];
  const float* Xn     = (const float*)d_in[2];
  const float* Xi     = (const float*)d_in[3];
  const float* Target = (const float*)d_in[4];
  const float* init_c = (const float*)d_in[5];
  const float* init_h = (const float*)d_in[6];
  const float* Wih    = (const float*)d_in[7];
  const float* Whh    = (const float*)d_in[8];
  const float* bih    = (const float*)d_in[9];
  const float* bhh    = (const float*)d_in[10];
  const float* Wc     = (const float*)d_in[11];
  const float* Uc     = (const float*)d_in[12];
  const float* bc     = (const float*)d_in[13];
  const float* Wg     = (const float*)d_in[14];
  const float* Ug     = (const float*)d_in[15];
  const float* bg     = (const float*)d_in[16];
  const float* Wf     = (const float*)d_in[17];
  const float* Uf     = (const float*)d_in[18];
  const float* bfv    = (const float*)d_in[19];
  const float* Wo     = (const float*)d_in[20];
  const float* Uo     = (const float*)d_in[21];
  const float* bov    = (const float*)d_in[22];
  const float* Wa     = (const float*)d_in[23];
  const float* Watt   = (const float*)d_in[24];
  const float* batt   = (const float*)d_in[25];
  const float* vatt   = (const float*)d_in[26];
  const float* W1     = (const float*)d_in[27];
  const float* b1     = (const float*)d_in[28];
  const float* W2     = (const float*)d_in[29];
  const float* b2     = (const float*)d_in[30];
  const float* W3     = (const float*)d_in[31];
  const float* b3     = (const float*)d_in[32];

  float* ws     = (float*)d_ws;
  float* result = ws;                                  // [B][T][4H] f32 (32 MB)
  float* Y2     = result + (size_t)BB * TT * G4;       // [B][T][H]  f32 (8 MB)
  float* scores = Y2 + (size_t)BB * TT * HH;           // [B][T]
  float* pad    = scores + (size_t)BB * TT;            // keep 16B alignment
  bhalf* WhhT = (bhalf*)(pad + 32);
  bhalf* WcT  = WhhT + (size_t)G4 * HH;
  bhalf* UcT  = WcT  + (size_t)4 * HH * HH;
  bhalf* WgT  = UcT  + (size_t)4 * HH * HH;
  bhalf* UgT  = WgT  + (size_t)4 * HH * HH;
  bhalf* WfT  = UgT  + (size_t)4 * HH * HH;
  bhalf* UfT  = WfT  + (size_t)G4 * HH;
  bhalf* WoT  = UfT  + (size_t)HH * HH;
  bhalf* UoT  = WoT  + (size_t)G4 * HH;
  bhalf* WaT  = UoT  + (size_t)HH * HH;

  zero_f32<<<2048, 256, 0, stream>>>(result, (size_t)BB * TT * G4);
  cvt_bf16<<<64, 256, 0, stream>>>(Whh, WhhT, (size_t)G4 * HH);   // Whh already [N][K]
  transpose_bf16<<<256, 256, 0, stream>>>(Wc, WcT, HH, HH, 4);
  transpose_bf16<<<256, 256, 0, stream>>>(Uc, UcT, HH, HH, 4);
  transpose_bf16<<<256, 256, 0, stream>>>(Wg, WgT, HH, HH, 4);
  transpose_bf16<<<256, 256, 0, stream>>>(Ug, UgT, HH, HH, 4);
  transpose_bf16<<<256, 256, 0, stream>>>(Wf, WfT, G4, HH, 1);
  transpose_bf16<<<64,  256, 0, stream>>>(Uf, UfT, HH, HH, 1);
  transpose_bf16<<<256, 256, 0, stream>>>(Wo, WoT, G4, HH, 1);
  transpose_bf16<<<64,  256, 0, stream>>>(Uo, UoT, HH, HH, 1);
  transpose_bf16<<<64,  256, 0, stream>>>(Wa, WaT, HH, HH, 1);

  stage1_lstm<<<34, 256, 0, stream>>>(Y, Xp, Xn, Xi, Wih, bih, bhh, WhhT, result);
  stage2_milstm<<<1, 256, 0, stream>>>(result, init_c, init_h,
                                       WcT, UcT, WgT, UgT, WfT, UfT, WoT, UoT, WaT,
                                       bc, bg, bfv, bov, Y2);
  stage3_scores<<<dim3(TT, BB), 128, 0, stream>>>(Y2, Watt, batt, vatt, scores);
  stage3_head<<<BB, 128, 0, stream>>>(scores, Y2, W1, b1, W2, b2, W3, b3, (float*)d_out);
  loss_kernel<<<1, BB, 0, stream>>>(Target, (float*)d_out);
}
